// WindowAttention_9277129359820
// MI455X (gfx1250) — matrix operand order repaired and checked by reading, not compiled, not run on hardware
//
#include <hip/hip_runtime.h>
#include <hip/hip_bf16.h>

typedef __attribute__((ext_vector_type(16))) _Float16 v16h;
typedef __attribute__((ext_vector_type(8)))  float    v8f;

#define TT      5
#define WS      8
#define HEADS   6
#define DIM     192
#define HDIM    32
#define NTOK    320          // T*WS*WS
#define SHIFT   4
#define SCALE   0.17677669529663687f   // 32^-0.5
#define NW      64           // windows per image (8x8)

// ---------------------------------------------------------------------------
// Kernel 1: per (window, head) attention.  grid = 128*6 = 768 blocks,
// 128 threads = 4 waves; each wave handles 5 query tiles of 16 rows.
// ---------------------------------------------------------------------------
__global__ __launch_bounds__(128) void WindowAttention_attn_kernel(
    const float* __restrict__ qkv,        // (2, 5*64*64, 576)
    const float* __restrict__ bias_table, // (1185, 6)
    const float* __restrict__ mask,       // (64, 320, 320)
    _Float16*    __restrict__ aout)       // (128, 320, 192) f16 scratch
{
    __shared__ __align__(32) _Float16 Ks[NTOK * HDIM];  // 20 KB
    __shared__ __align__(32) _Float16 Vs[NTOK * HDIM];  // 20 KB
    __shared__ __align__(32) _Float16 patch[4][16 * 32]; // 4 KB (per-wave)

    const int bh = blockIdx.x;
    const int b_ = bh / HEADS;
    const int h  = bh % HEADS;
    const int b  = b_ >> 6;           // image
    const int wh = (b_ >> 3) & 7;     // window row
    const int ww = b_ & 7;            // window col
    const int wmask = b_ & 63;        // mask window index

    const int tid  = threadIdx.x;
    const int lane = tid & 31;
    const int wid  = tid >> 5;
    const int hi   = lane >> 4;       // half-wave
    const int ln   = lane & 15;

    // ---- stage K and V (this head) into LDS as f16, coalesced over d ----
    for (int i = tid; i < NTOK * HDIM; i += 128) {
        int n = i >> 5, d = i & 31;
        int t = n >> 6, p = n & 63, ph = p >> 3, pw = p & 7;
        int row = (wh * 8 + ph + SHIFT) & 63;
        int col = (ww * 8 + pw + SHIFT) & 63;
        long base = ((long)(b * TT + t) * 4096 + row * 64 + col) * (3 * DIM);
        Ks[i] = (_Float16)qkv[base + DIM     + h * HDIM + d];
        Vs[i] = (_Float16)qkv[base + 2 * DIM + h * HDIM + d];
    }
    __syncthreads();

    _Float16* pp = &patch[wid][0];

    for (int qi = 0; qi < 5; ++qi) {
        const int qt = qi * 4 + wid;        // query tile 0..19

        // ---- load Q tile straight into WMMA A layout (scaled) ----
        {
            int qtok = qt * 16 + ln;
            int t = qtok >> 6, p = qtok & 63, ph = p >> 3, pw = p & 7;
            int row = (wh * 8 + ph + SHIFT) & 63;
            int col = (ww * 8 + pw + SHIFT) & 63;
            const float* qp = qkv + ((long)(b * TT + t) * 4096 + row * 64 + col) * (3 * DIM)
                                  + h * HDIM;
            v16h a;
            #pragma unroll
            for (int j = 0; j < 8; ++j) a[j]     = (_Float16)(qp[hi * 16 + j] * SCALE);
            #pragma unroll
            for (int j = 0; j < 8; ++j) a[8 + j] = (_Float16)(qp[hi * 16 + 8 + j] * SCALE);

            // ---- scores: 20 WMMAs (Q 16x32  x  K^T 32x16) ----
            v8f acc[20];
            #pragma unroll
            for (int kt = 0; kt < 20; ++kt) {
                v16h bk = *(const v16h*)(&Ks[(kt * 16 + ln) * HDIM + hi * 16]);
                v8f z = (v8f){0.f,0.f,0.f,0.f,0.f,0.f,0.f,0.f};
                acc[kt] = __builtin_amdgcn_wmma_f32_16x16x32_f16(
                    false, a, false, bk, (short)0, z, false, false);
            }

            // ---- add relative-position bias (recomputed) + shift mask ----
            int qt_t[8], qt_h[8], qt_w[8];
            #pragma unroll
            for (int r = 0; r < 8; ++r) {
                int q = qt * 16 + hi * 8 + r;
                qt_t[r] = q >> 6; int pq = q & 63;
                qt_h[r] = pq >> 3; qt_w[r] = pq & 7;
            }
            #pragma unroll
            for (int kt = 0; kt < 20; ++kt) {
                int k = kt * 16 + ln;
                int tk = k >> 6, pk = k & 63, hk = pk >> 3, wk = pk & 7;
                #pragma unroll
                for (int r = 0; r < 8; ++r) {
                    int q  = qt * 16 + hi * 8 + r;
                    int dt = qt_t[r] - tk, dh = qt_h[r] - hk, dw = qt_w[r] - wk;
                    int idx = (dt + 7) * 15 + (dh + 7) + dw;
                    float bv = bias_table[idx * HEADS + h];
                    float mv = mask[((long)wmask * NTOK + q) * NTOK + k];
                    acc[kt][r] += bv + mv;
                }
            }

            // ---- softmax over 320 keys: reg-local + 16-lane butterfly ----
            float rinv[8];
            #pragma unroll
            for (int r = 0; r < 8; ++r) {
                float m = -1e30f;
                #pragma unroll
                for (int kt = 0; kt < 20; ++kt) m = fmaxf(m, acc[kt][r]);
                #pragma unroll
                for (int s = 1; s < 16; s <<= 1) m = fmaxf(m, __shfl_xor(m, s, 32));
                float sum = 0.f;
                #pragma unroll
                for (int kt = 0; kt < 20; ++kt) {
                    float e = __expf(acc[kt][r] - m);
                    acc[kt][r] = e;
                    sum += e;
                }
                #pragma unroll
                for (int s = 1; s < 16; s <<= 1) sum += __shfl_xor(sum, s, 32);
                rinv[r] = 1.0f / sum;
            }

            // ---- out = P x V : 10 K-chunks of 32, 2 output col tiles ----
            v8f o0 = (v8f){0.f,0.f,0.f,0.f,0.f,0.f,0.f,0.f};
            v8f o1 = (v8f){0.f,0.f,0.f,0.f,0.f,0.f,0.f,0.f};
            for (int c = 0; c < 10; ++c) {
                // transpose P chunk D-layout -> A-layout through per-wave LDS
                #pragma unroll
                for (int k2 = 0; k2 < 2; ++k2) {
                    int kt = 2 * c + k2;
                    #pragma unroll
                    for (int r = 0; r < 8; ++r)
                        pp[(hi * 8 + r) * 32 + k2 * 16 + ln] =
                            (_Float16)(acc[kt][r] * rinv[r]);
                }
                v16h pa;
                #pragma unroll
                for (int j = 0; j < 8; ++j) pa[j]     = pp[ln * 32 + hi * 8 + j];
                #pragma unroll
                for (int j = 0; j < 8; ++j) pa[8 + j] = pp[ln * 32 + 16 + hi * 8 + j];

                v16h b0, b1;
                #pragma unroll
                for (int j = 0; j < 16; ++j) {
                    int key = 32 * c + hi * 8 + j + (j >= 8 ? 8 : 0);
                    b0[j] = Vs[key * HDIM + ln];
                    b1[j] = Vs[key * HDIM + 16 + ln];
                }
                o0 = __builtin_amdgcn_wmma_f32_16x16x32_f16(
                        false, pa, false, b0, (short)0, o0, false, false);
                o1 = __builtin_amdgcn_wmma_f32_16x16x32_f16(
                        false, pa, false, b1, (short)0, o1, false, false);
            }

            // ---- store per-head attention output (f16 scratch) ----
            #pragma unroll
            for (int r = 0; r < 8; ++r) {
                int token = qt * 16 + hi * 8 + r;
                _Float16* op = aout + ((long)b_ * NTOK + token) * DIM + h * HDIM;
                op[ln]      = (_Float16)o0[r];
                op[16 + ln] = (_Float16)o1[r];
            }
        }
    }
}

// ---------------------------------------------------------------------------
// Kernel 2: projection (40960 x 192) @ proj_w^T + proj_b, fused with
// window-reverse + cyclic un-shift.  grid = 640 blocks x 4 waves,
// one 16x192 output tile per wave.
// ---------------------------------------------------------------------------
__global__ __launch_bounds__(128) void WindowAttention_proj_kernel(
    const _Float16* __restrict__ aout,    // (128, 320, 192) f16
    const float*    __restrict__ proj_w,  // (192, 192)
    const float*    __restrict__ proj_b,  // (192,)
    float*          __restrict__ out)     // (2, 5*64*64, 192)
{
    const int tid  = threadIdx.x;
    const int lane = tid & 31;
    const int wid  = tid >> 5;
    const int hi   = lane >> 4;
    const int ln   = lane & 15;

    const int mt    = blockIdx.x * 4 + wid;   // 0..2559 row tiles
    const int brow  = mt * 16;                // first global row
    const int b_    = brow / NTOK;
    const int ntok0 = brow % NTOK;

    v8f acc[12];
    #pragma unroll
    for (int jt = 0; jt < 12; ++jt)
        acc[jt] = (v8f){0.f,0.f,0.f,0.f,0.f,0.f,0.f,0.f};

    for (int kt = 0; kt < 6; ++kt) {
        const _Float16* ap = aout + ((long)(brow + ln)) * DIM + kt * 32;
        v16h a;
        #pragma unroll
        for (int j = 0; j < 8; ++j) a[j]     = ap[hi * 8 + j];
        #pragma unroll
        for (int j = 0; j < 8; ++j) a[8 + j] = ap[16 + hi * 8 + j];

        #pragma unroll
        for (int jt = 0; jt < 12; ++jt) {
            // B[d][col]: col = jt*16+ln, d = kt*32 + hi*8 + j (+8 for j>=8) ; out = x @ W^T
            const float* wp = proj_w + (jt * 16 + ln) * DIM + kt * 32 + hi * 8;
            v16h bw;
            #pragma unroll
            for (int j = 0; j < 16; ++j) bw[j] = (_Float16)wp[j + (j >= 8 ? 8 : 0)];
            acc[jt] = __builtin_amdgcn_wmma_f32_16x16x32_f16(
                         false, a, false, bw, (short)0, acc[jt], false, false);
        }
    }

    const int bb = b_ >> 6, wh = (b_ >> 3) & 7, ww = b_ & 7;
    #pragma unroll
    for (int jt = 0; jt < 12; ++jt) {
        float bias = proj_b[jt * 16 + ln];
        #pragma unroll
        for (int r = 0; r < 8; ++r) {
            int n = ntok0 + hi * 8 + r;
            int t = n >> 6, p = n & 63, ph = p >> 3, pw = p & 7;
            int row = (wh * 8 + ph + SHIFT) & 63;   // un-shift (roll +4)
            int col = (ww * 8 + pw + SHIFT) & 63;
            long dst = ((long)(bb * TT + t) * 4096 + row * 64 + col) * DIM
                       + jt * 16 + ln;
            out[dst] = acc[jt][r] + bias;
        }
    }
}

// ---------------------------------------------------------------------------
extern "C" void kernel_launch(void* const* d_in, const int* in_sizes, int n_in,
                              void* d_out, int out_size, void* d_ws, size_t ws_size,
                              hipStream_t stream) {
    const float* qkv        = (const float*)d_in[0];
    const float* bias_table = (const float*)d_in[1];
    const float* proj_w     = (const float*)d_in[2];
    const float* proj_b     = (const float*)d_in[3];
    const float* mask       = (const float*)d_in[4];
    float*       out        = (float*)d_out;
    _Float16*    aout       = (_Float16*)d_ws;   // 128*320*192 f16 = 15.7 MB

    WindowAttention_attn_kernel<<<128 * HEADS, 128, 0, stream>>>(
        qkv, bias_table, mask, aout);
    WindowAttention_proj_kernel<<<2560 / 4, 128, 0, stream>>>(
        aout, proj_w, proj_b, out);
}
